// Attention_5987184411180
// MI455X (gfx1250) — matrix operand order repaired and checked by reading, not compiled, not run on hardware
//
#include <hip/hip_runtime.h>

// ---------------- types / WMMA helpers ----------------
typedef __attribute__((ext_vector_type(16))) __bf16 v16bf;
typedef __attribute__((ext_vector_type(8)))  float  v8f;
typedef __attribute__((ext_vector_type(4)))  unsigned int u32x4;
typedef __attribute__((ext_vector_type(8)))  int    i32x8;
typedef __attribute__((ext_vector_type(4)))  int    i32x4;

#define WMMA_BF16(a, b, c) \
  __builtin_amdgcn_wmma_f32_16x16x32_bf16(false, (a), false, (b), (short)0, (c), false, false)

union FragBF { uint4 u[2]; v16bf v; };

// A-matrix 16x32 bf16 fragment: lanes 0-15 -> M=lane, K=k0+0..15 contiguous;
// lanes 16-31 -> M=lane-16, K=k0+16..31. Caller supplies per-lane row pointer.
__device__ __forceinline__ v16bf load_a_frag(const __bf16* row, int k0, int lane) {
  int ko = k0 + ((lane & 16) ? 16 : 0);
  FragBF f;
  f.u[0] = *(const uint4*)(row + ko);        // 8 halves: K ko..ko+7
  f.u[1] = *(const uint4*)(row + ko + 8);    // 8 halves: K ko+8..ko+15
  return f.v;
}

// B-matrix 32x16 bf16 fragment: lane = N column; lanes 0-15 hold K=k0..k0+15 contiguous,
// lanes 16-31 hold K=k0+16..k0+31. Caller supplies per-lane column pointer (K-contiguous).
__device__ __forceinline__ v16bf load_b_frag(const __bf16* col, int k0, int lane) {
  int ko = k0 + ((lane & 16) ? 16 : 0);
  FragBF f;
  f.u[0] = *(const uint4*)(col + ko);
  f.u[1] = *(const uint4*)(col + ko + 8);
  return f.v;
}

// Same B-fragment shape but sourced from an LDS-staged tile: sB[64 cols][32 K halves].
__device__ __forceinline__ v16bf load_b_frag_lds(const unsigned short* sB, int j, int lane) {
  const unsigned short* bp = sB + (j * 16 + (lane & 15)) * 32 + ((lane & 16) ? 16 : 0);
  FragBF f;
  f.u[0] = *(const uint4*)bp;
  f.u[1] = *(const uint4*)(bp + 8);
  return f.v;
}

__device__ __forceinline__ unsigned short f2bfbits(float f) {
  __bf16 h = (__bf16)f;
  return __builtin_bit_cast(unsigned short, h);
}

// ---------------- problem constants ----------------
#define DMODEL 2048
#define SEQ    2048
#define NB     2
#define NH     16
#define DH     128
#define MTOT   (NB * SEQ)        // 4096
#define NQKV   (3 * DMODEL)      // 6144

// ---------------- Tensor Data Mover: 2-D bf16 tile (32 K-elems x 64 rows) -> LDS ----------------
#if __has_builtin(__builtin_amdgcn_tensor_load_to_lds)
#define HAVE_TDM 1
#else
#define HAVE_TDM 0
#endif

#if HAVE_TDM
// D# packing per CDNA5 ISA:
//  group0: [1:0]=count=1, [63:32]=lds_addr, [120:64]=global_addr, [127:126]=type=2
//  group1: [17:16]=data_size(1 -> 2B), [79:48]=tensor_dim0, [111:80]=tensor_dim1,
//          [127:112]=tile_dim0(=32), [143:128]=tile_dim1(=64), [207:160]=tensor_dim0_stride
__device__ __forceinline__ void tdm_load_b_tile(unsigned lds_off, const __bf16* gsrc,
                                                unsigned K, unsigned Ntot) {
  unsigned long long ga = (unsigned long long)(const void*)gsrc;
  u32x4 g0;
  g0[0] = 1u;                                                  // count=1
  g0[1] = lds_off;                                             // lds_addr (bytes)
  g0[2] = (unsigned)(ga & 0xffffffffu);                        // global_addr lo
  g0[3] = (unsigned)((ga >> 32) & 0x1ffffffu) | (2u << 30);    // global_addr hi | type=2
  i32x8 g1;
  g1[0] = (int)(1u << 16);                                     // data_size = 2 bytes
  g1[1] = (int)((K & 0xffffu) << 16);                          // tensor_dim0 lo16
  g1[2] = (int)((K >> 16) | ((Ntot & 0xffffu) << 16));         // td0 hi16 | td1 lo16
  g1[3] = (int)((Ntot >> 16) | (32u << 16));                   // td1 hi16 | tile_dim0=32
  g1[4] = (int)64u;                                            // tile_dim1=64, tile_dim2=0
  g1[5] = (int)K;                                              // tensor_dim0_stride lo32
  g1[6] = 0;                                                   // td0_stride hi16 | td1_stride lo16
  g1[7] = 0;
  i32x4 z4 = {0, 0, 0, 0};
#if __clang_major__ >= 23
  i32x8 z8 = {0, 0, 0, 0, 0, 0, 0, 0};
  __builtin_amdgcn_tensor_load_to_lds(g0, g1, z4, z4, z8, 0);
#else
  __builtin_amdgcn_tensor_load_to_lds(g0, g1, z4, z4, 0);
#endif
}
#endif

// Fallback cooperative staging (also used if TDM builtin absent): 128 threads copy 64x32 halves.
__device__ __forceinline__ void stage_b_tile_coop(unsigned short* dst, const __bf16* src,
                                                  int K, int tid) {
  int row = tid >> 1, half = (tid & 1) * 16;
  const uint4* s = (const uint4*)(src + (size_t)row * K + half);
  uint4* d = (uint4*)(dst + row * 32 + half);
  d[0] = s[0];
  d[1] = s[1];
}

// ---------------- kernel 1: fp32 -> bf16 convert ----------------
__global__ void k_cvt_bf16(const float* __restrict__ src, __bf16* __restrict__ dst, int n) {
  int i = blockIdx.x * 256 + threadIdx.x;
  if (i < n) dst[i] = (__bf16)src[i];
}

// ---------------- GEMM tile body: C(64x64) = A(64xK) * B(NxK)^T, LDS/TDM staged B ----------------
// Returns 4 accumulators per wave (wave owns rows tileM..tileM+15, all 64 N columns).
__device__ __forceinline__ void gemm_tile_body(
    const __bf16* __restrict__ A, const __bf16* __restrict__ B,
    int K, unsigned Ntot, int tileM, int tileN,
    int lane, int wave, int tid,
    unsigned short (*sB)[64 * 32], v8f acc[4]) {
  const __bf16* arow = A + (size_t)(tileM + (lane & 15)) * K;
  const int NIT = K / 32;

  // prologue: stage tile 0
#if HAVE_TDM
  if (wave == 0) {
    tdm_load_b_tile((unsigned)(size_t)&sB[0][0], B + (size_t)tileN * K, (unsigned)K, Ntot);
    __builtin_amdgcn_s_wait_tensorcnt(0);
  }
#else
  stage_b_tile_coop(&sB[0][0], B + (size_t)tileN * K, K, tid);
#endif
  __syncthreads();

  v16bf af = load_a_frag(arow, 0, lane);
  for (int it = 0; it < NIT; ++it) {
    int cur = it & 1;
    int k0 = it * 32;
    // kick off next tile's DMA while this tile computes
    if (it + 1 < NIT) {
#if HAVE_TDM
      if (wave == 0)
        tdm_load_b_tile((unsigned)(size_t)&sB[cur ^ 1][0],
                        B + (size_t)tileN * K + (k0 + 32), (unsigned)K, Ntot);
#else
      stage_b_tile_coop(&sB[cur ^ 1][0], B + (size_t)tileN * K + (k0 + 32), K, tid);
#endif
    }
    __builtin_prefetch(arow + k0 + 128, 0, 1);          // global_prefetch_b8
    v16bf af_n = af;
    if (it + 1 < NIT) af_n = load_a_frag(arow, k0 + 32, lane);   // register double-buffer

    v16bf bf0 = load_b_frag_lds(&sB[cur][0], 0, lane);
    v16bf bf1 = load_b_frag_lds(&sB[cur][0], 1, lane);
    v16bf bf2 = load_b_frag_lds(&sB[cur][0], 2, lane);
    v16bf bf3 = load_b_frag_lds(&sB[cur][0], 3, lane);
    acc[0] = WMMA_BF16(af, bf0, acc[0]);
    acc[1] = WMMA_BF16(af, bf1, acc[1]);
    acc[2] = WMMA_BF16(af, bf2, acc[2]);
    acc[3] = WMMA_BF16(af, bf3, acc[3]);

#if HAVE_TDM
    if (wave == 0) __builtin_amdgcn_s_wait_tensorcnt(0);
#endif
    __syncthreads();
    af = af_n;
  }
}

// ---------------- kernel 2: QKV GEMM, routed to q/k/v ----------------
__global__ __launch_bounds__(128) void k_qkv_gemm(
    const __bf16* __restrict__ A, const __bf16* __restrict__ B,
    const float* __restrict__ bias,
    float* __restrict__ qout, float* __restrict__ kout, float* __restrict__ vout) {
  __shared__ unsigned short sB[2][64 * 32];
  int lane = threadIdx.x & 31, wave = threadIdx.x >> 5;
  int tileM = blockIdx.y * 64 + wave * 16;
  int tileN = blockIdx.x * 64;

  v8f acc[4] = {};
  gemm_tile_body(A, B, DMODEL, NQKV, tileM, tileN, lane, wave, threadIdx.x, sB, acc);

  // route: n<2048 -> q (ws, fp32), 2048..4095 -> k_new (d_out), 4096.. -> v_new (d_out)
  int seg = tileN >> 11;
  float* dst = (seg == 0) ? qout : ((seg == 1) ? kout : vout);
  int colbase = (tileN & 2047);
#pragma unroll
  for (int j = 0; j < 4; ++j) {
    int col = colbase + j * 16 + (lane & 15);
    float bv = bias[tileN + j * 16 + (lane & 15)];
#pragma unroll
    for (int r = 0; r < 8; ++r) {
      int row = tileM + r + ((lane & 16) ? 8 : 0);
      dst[(size_t)row * DMODEL + col] = acc[j][r] + bv;
    }
  }
}

// ---------------- kernel 3: per-head RMS norm of q,k -> bf16; V -> transposed bf16 ----------------
__global__ __launch_bounds__(128) void k_headnorm(
    const float* __restrict__ qraw, const float* __restrict__ kraw, const float* __restrict__ vraw,
    const float* __restrict__ wq, const float* __restrict__ wk,
    __bf16* __restrict__ qn, __bf16* __restrict__ kn, __bf16* __restrict__ vT) {
  int t = threadIdx.x;
  int bx = blockIdx.x;
  int m = bx >> 4, h = bx & 15;
  int b = m >> 11, s = m & 2047;
  size_t base = (size_t)m * DMODEL + h * DH;

  float q = qraw[base + t];
  float k = kraw[base + t];
  float v = vraw[base + t];

  float sq = q * q, sk = k * k;
#pragma unroll
  for (int off = 1; off < 32; off <<= 1) {
    sq += __shfl_xor(sq, off, 32);
    sk += __shfl_xor(sk, off, 32);
  }
  __shared__ float red[8];
  int lane = t & 31, wave = t >> 5;
  if (lane == 0) { red[wave] = sq; red[4 + wave] = sk; }
  __syncthreads();
  float totq = red[0] + red[1] + red[2] + red[3];
  float totk = red[4] + red[5] + red[6] + red[7];
  float rq = rsqrtf(totq * (1.0f / DH) + 1e-6f);
  float rk = rsqrtf(totk * (1.0f / DH) + 1e-6f);

  qn[base + t] = (__bf16)(q * rq * wq[t]);
  kn[base + t] = (__bf16)(k * rk * wk[t]);
  // V^T layout: vT[((b*16+h)*128 + d) * 2048 + s]  (keys contiguous for PV B-fragments)
  vT[((size_t)(b * NH + h) * DH + t) * SEQ + s] = (__bf16)v;
}

// ---------------- kernel 4: flash attention, scale=1.0, non-causal ----------------
// grid = NB*NH*(SEQ/64); 128 threads = 4 waves; each wave owns 16 query rows.
__global__ __launch_bounds__(128) void k_attn(
    const __bf16* __restrict__ qn, const __bf16* __restrict__ kn,
    const __bf16* __restrict__ vT, __bf16* __restrict__ zout) {
  int lane = threadIdx.x & 31, wave = threadIdx.x >> 5;
  int bx = blockIdx.x;
  int qb = bx & 31, h = (bx >> 5) & 15, b = bx >> 9;
  int q0 = qb * 64 + wave * 16;           // query row within sequence
  int m0 = b * SEQ;                        // token base

  // Q fragments (resident): 16 rows x d_head=128 -> 4 K-steps of 32
  const __bf16* qrow = qn + (size_t)(m0 + q0 + (lane & 15)) * DMODEL + h * DH;
  v16bf qf[4];
#pragma unroll
  for (int kk = 0; kk < 4; ++kk) qf[kk] = load_a_frag(qrow, kk * 32, lane);

  v8f z[8] = {};
  float run_max[8], run_sum[8];
#pragma unroll
  for (int r = 0; r < 8; ++r) { run_max[r] = -3.0e38f; run_sum[r] = 0.0f; }

  __shared__ unsigned short sP[4][16 * 32];   // per-wave P staging (C-layout -> A-layout)
  const __bf16* kbase = kn + (size_t)m0 * DMODEL + h * DH;
  const __bf16* vbase = vT + (size_t)(b * NH + h) * DH * SEQ;

  for (int kt = 0; kt < SEQ; kt += 32) {
    // ---- S = Qn * Kn^T for 16 queries x 32 keys: batch all loads, then WMMA chain ----
    const __bf16* krow_a = kbase + (size_t)(kt + (lane & 15)) * DMODEL;
    const __bf16* krow_b = kbase + (size_t)(kt + 16 + (lane & 15)) * DMODEL;
    v16bf kfa[4], kfb[4];
#pragma unroll
    for (int kk = 0; kk < 4; ++kk) {
      kfa[kk] = load_b_frag(krow_a, kk * 32, lane);
      kfb[kk] = load_b_frag(krow_b, kk * 32, lane);
    }
    v8f sa = {}, sb = {};
#pragma unroll
    for (int kk = 0; kk < 4; ++kk) {
      sa = WMMA_BF16(qf[kk], kfa[kk], sa);
      sb = WMMA_BF16(qf[kk], kfb[kk], sb);
    }

    // ---- online softmax. Acc layout: element r, lane l -> row r+8*(l>=16), col l&15 ----
    float alpha[8];
#pragma unroll
    for (int r = 0; r < 8; ++r) {
      float mx = fmaxf(sa[r], sb[r]);
#pragma unroll
      for (int off = 1; off < 16; off <<= 1) mx = fmaxf(mx, __shfl_xor(mx, off, 32));
      float nm = fmaxf(run_max[r], mx);
      float al = __expf(run_max[r] - nm);
      float pa = __expf(sa[r] - nm);
      float pb = __expf(sb[r] - nm);
      float rs = pa + pb;
#pragma unroll
      for (int off = 1; off < 16; off <<= 1) rs += __shfl_xor(rs, off, 32);
      run_sum[r] = run_sum[r] * al + rs;
      run_max[r] = nm;
      alpha[r] = al;
      int row = r + ((lane & 16) ? 8 : 0);
      int key = lane & 15;
      sP[wave][row * 32 + key]      = f2bfbits(pa);
      sP[wave][row * 32 + key + 16] = f2bfbits(pb);
    }

    // rescale running z by alpha (same row mapping as accumulators)
#pragma unroll
    for (int j = 0; j < 8; ++j)
#pragma unroll
      for (int r = 0; r < 8; ++r) z[j][r] *= alpha[r];

    // ---- PV: reload P in A-layout from LDS; batch V loads, then WMMA chain ----
    {
      int prow = lane & 15;
      int pko = (lane & 16) ? 16 : 0;
      FragBF pf;
      pf.u[0] = *(const uint4*)&sP[wave][prow * 32 + pko];
      pf.u[1] = *(const uint4*)&sP[wave][prow * 32 + pko + 8];
      v16bf vf[8];
#pragma unroll
      for (int j = 0; j < 8; ++j) {
        const __bf16* vrow = vbase + (size_t)(j * 16 + (lane & 15)) * SEQ + kt;
        vf[j] = load_b_frag(vrow, 0, lane);
      }
#pragma unroll
      for (int j = 0; j < 8; ++j) z[j] = WMMA_BF16(pf.v, vf[j], z[j]);
    }
  }

  // finalize: z / run_sum, store bf16 in [token][h*128+d] layout for O-GEMM
  float inv[8];
#pragma unroll
  for (int r = 0; r < 8; ++r) inv[r] = 1.0f / run_sum[r];
#pragma unroll
  for (int j = 0; j < 8; ++j) {
#pragma unroll
    for (int r = 0; r < 8; ++r) {
      int row = m0 + q0 + r + ((lane & 16) ? 8 : 0);
      int col = h * DH + j * 16 + (lane & 15);
      zout[(size_t)row * DMODEL + col] = (__bf16)(z[j][r] * inv[r]);
    }
  }
}

// ---------------- kernel 5: output GEMM: out = z * Wo^T + b_o ----------------
__global__ __launch_bounds__(128) void k_out_gemm(
    const __bf16* __restrict__ A, const __bf16* __restrict__ B,
    const float* __restrict__ bias, float* __restrict__ C) {
  __shared__ unsigned short sB[2][64 * 32];
  int lane = threadIdx.x & 31, wave = threadIdx.x >> 5;
  int tileM = blockIdx.y * 64 + wave * 16;
  int tileN = blockIdx.x * 64;

  v8f acc[4] = {};
  gemm_tile_body(A, B, DMODEL, DMODEL, tileM, tileN, lane, wave, threadIdx.x, sB, acc);

#pragma unroll
  for (int j = 0; j < 4; ++j) {
    int col = tileN + j * 16 + (lane & 15);
    float bv = bias[col];
#pragma unroll
    for (int r = 0; r < 8; ++r) {
      int row = tileM + r + ((lane & 16) ? 8 : 0);
      C[(size_t)row * DMODEL + col] = acc[j][r] + bv;
    }
  }
}

// ---------------- host launcher ----------------
extern "C" void kernel_launch(void* const* d_in, const int* in_sizes, int n_in,
                              void* d_out, int out_size, void* d_ws, size_t ws_size,
                              hipStream_t stream) {
  const float* x     = (const float*)d_in[0];
  const float* W_qkv = (const float*)d_in[1];
  const float* b_qkv = (const float*)d_in[2];
  const float* W_o   = (const float*)d_in[3];
  const float* b_o   = (const float*)d_in[4];
  const float* wq    = (const float*)d_in[5];
  const float* wk    = (const float*)d_in[6];

  float* out   = (float*)d_out;                        // [4096 x 2048]
  float* k_out = out + (size_t)MTOT * DMODEL;          // k_new  [4096 x 2048]
  float* v_out = k_out + (size_t)MTOT * DMODEL;        // v_new  [4096 x 2048]

  // workspace layout (all 256B-aligned by construction)
  char* ws = (char*)d_ws;
  size_t off = 0;
  __bf16* x_bf    = (__bf16*)(ws + off); off += (size_t)MTOT * DMODEL * 2;     // 16.8 MB
  __bf16* wqkv_bf = (__bf16*)(ws + off); off += (size_t)NQKV * DMODEL * 2;     // 25.2 MB
  __bf16* wo_bf   = (__bf16*)(ws + off); off += (size_t)DMODEL * DMODEL * 2;   //  8.4 MB
  float*  q_raw   = (float*) (ws + off); off += (size_t)MTOT * DMODEL * 4;     // 33.6 MB
  __bf16* qn_bf   = (__bf16*)(ws + off); off += (size_t)MTOT * DMODEL * 2;
  __bf16* kn_bf   = (__bf16*)(ws + off); off += (size_t)MTOT * DMODEL * 2;
  __bf16* vT_bf   = (__bf16*)(ws + off); off += (size_t)MTOT * DMODEL * 2;
  __bf16* z_bf    = (__bf16*)(ws + off); off += (size_t)MTOT * DMODEL * 2;     // total ~151 MB

  // 1) convert operands to bf16
  {
    int n1 = MTOT * DMODEL, n2 = NQKV * DMODEL, n3 = DMODEL * DMODEL;
    k_cvt_bf16<<<(n1 + 255) / 256, 256, 0, stream>>>(x, x_bf, n1);
    k_cvt_bf16<<<(n2 + 255) / 256, 256, 0, stream>>>(W_qkv, wqkv_bf, n2);
    k_cvt_bf16<<<(n3 + 255) / 256, 256, 0, stream>>>(W_o, wo_bf, n3);
  }

  // 2) QKV GEMM (writes k_new / v_new straight into d_out)
  {
    dim3 grid(NQKV / 64, MTOT / 64);
    k_qkv_gemm<<<grid, 128, 0, stream>>>(x_bf, wqkv_bf, b_qkv, q_raw, k_out, v_out);
  }

  // 3) head-wise RMS norm + V transpose
  k_headnorm<<<MTOT * NH, 128, 0, stream>>>(q_raw, k_out, v_out, wq, wk, qn_bf, kn_bf, vT_bf);

  // 4) attention (flash, scale=1.0, non-causal)
  k_attn<<<NB * NH * (SEQ / 64), 128, 0, stream>>>(qn_bf, kn_bf, vT_bf, z_bf);

  // 5) output projection
  {
    dim3 grid(DMODEL / 64, MTOT / 64);
    k_out_gemm<<<grid, 128, 0, stream>>>(z_bf, wo_bf, b_o, out);
  }
}
